// CTLSTMCell_90065464197524
// MI455X (gfx1250) — compile-verified, matrix operand order checked by reading, not run on hardware
//
#include <hip/hip_runtime.h>
#include <hip/hip_bf16.h>

// ---------------------------------------------------------------------------
// CT-LSTM cell, MI455X (gfx1250).
// g = [x|ht] @ [Wx|Wh]^T + (bx+bh)  -> 7 gates -> elementwise epilogue.
// bf16 WMMA (16x16x32) GEMM fused with the gate epilogue.
// Async global->LDS staging (ASYNCcnt), triple-buffered, 1 barrier/iter.
// Each wave: M=32 rows x 16 cols x 7 gates -> every B fragment feeds 2 WMMAs.
// k-loop rolled (unroll 1); gate chain manually software-pipelined with two
// named B-fragment registers so ds loads run 2 WMMA-pairs ahead.
// ---------------------------------------------------------------------------

typedef __attribute__((ext_vector_type(16))) __bf16 v16bf;
typedef __attribute__((ext_vector_type(8)))  float  v8f;

#define B_SZ 16384
#define H_SZ 512
#define K_SZ 1024           // I + H
#define NG   7
#define KT   32             // number of K-tiles (K_SZ / 32)
#define TILE_BYTES 24576    // 16 KB A tile (256 rows) + 8 KB B tile (8 gates)
#define NBUF 3

// workspace layout (bytes)
#define XC_OFF 0u                    // bf16 [B][1024]    = 33,554,432 B
#define WC_OFF 33554432u             // bf16 [3584][1024] =  7,340,032 B
#define BS_OFF 40894464u             // f32  [3584]       =     14,336 B

// ---------------------------------------------------------------------------
// Pack: dst[row][0:512] = a[row], dst[row][512:1024] = b[row], fp32 -> bf16
// ---------------------------------------------------------------------------
__global__ __launch_bounds__(256) void pack_concat_bf16(
    const float* __restrict__ a, const float* __restrict__ b,
    __bf16* __restrict__ dst, int rows)
{
    int idx4 = (blockIdx.x * 256 + threadIdx.x) * 4;
    if (idx4 >= rows * 1024) return;
    int row = idx4 >> 10;
    int col = idx4 & 1023;
    const float* src = (col < 512) ? (a + (size_t)row * 512 + col)
                                   : (b + (size_t)row * 512 + (col - 512));
    float4 v = *(const float4*)src;
    union { __bf16 h[4]; uint2 u; } r;
    r.h[0] = (__bf16)v.x; r.h[1] = (__bf16)v.y;
    r.h[2] = (__bf16)v.z; r.h[3] = (__bf16)v.w;
    *(uint2*)(dst + idx4) = r.u;
}

__global__ __launch_bounds__(256) void bias_sum(
    const float* __restrict__ bx, const float* __restrict__ bh,
    float* __restrict__ bs)
{
    int i = blockIdx.x * 256 + threadIdx.x;
    if (i < NG * H_SZ) bs[i] = bx[i] + bh[i];
}

// ---------------------------------------------------------------------------
__device__ __forceinline__ float sigmoidf_(float x) {
    return 1.0f / (1.0f + __expf(-x));
}

// Async DMA global->LDS (gfx1250, ASYNCcnt). INST_OFFSET adds to BOTH the
// global address and the LDS destination, so one pointer covers a contiguous
// per-thread slice.
#define ASYNC_B128(lds_off, ptr, imm)                                      \
    asm volatile("global_load_async_to_lds_b128 %0, %1, off offset:" imm   \
                 :: "v"(lds_off), "v"(ptr) : "memory")

// ---------------------------------------------------------------------------
// Fused GEMM (bf16 WMMA) + LSTM epilogue.
// grid = (B/256, H/16), block = 256 (8 waves). Wave w owns rows w*32..w*32+31.
// ---------------------------------------------------------------------------
__global__ __launch_bounds__(256) void ctlstm_wmma(
    const __bf16* __restrict__ Xc,   // [B][1024]
    const __bf16* __restrict__ Wc,   // [3584][1024]
    const float*  __restrict__ bs,   // [3584]
    const float*  __restrict__ ct,   // [B][512]
    float*        __restrict__ out)  // h | c | c_bar | o | decay, each [B][512]
{
    __shared__ uint4 lds4[NBUF * TILE_BYTES / 16];

    const int tid  = threadIdx.x;
    const int wave = tid >> 5;
    const int lane = tid & 31;
    const int lg   = lane >> 4;
    const int ln   = lane & 15;

    const int rowBase = blockIdx.x * 256;
    const int colBase = blockIdx.y * 16;

    // ---- per-thread staging: 6 full-EXEC async b128 per tile ----
    // A: thread t copies the 64-B k-slice of row t (offsets 0/16/32/48).
    const char* gA = (const char*)(Xc + (size_t)(rowBase + tid) * K_SZ);
    const unsigned aDst = (unsigned)tid * 64u;
    // B: 8 gate sub-tiles x 16 cols (gate 7 duplicates gate 6 to keep the
    // async-instruction count uniform); thread t copies 32 B of row t/2.
    const int sRowB = tid >> 1;                       // 0..127
    int bg = sRowB >> 4; if (bg > 6) bg = 6;
    const int bn = sRowB & 15;
    const char* gB = (const char*)(Wc + (size_t)(bg * H_SZ + colBase + bn) * K_SZ)
                   + (tid & 1) * 32;
    const unsigned bDst = 16384u + (unsigned)sRowB * 64u + (tid & 1) * 32u;

    const unsigned ldsBase = (unsigned)(unsigned long long)(void*)lds4;

    // ---- per-wave compute fragment offsets (documented gfx1250 layouts:
    // lanes 0-15: K 0..7 / 16..23, lanes 16-31: K 8..15 / 24..31) ----
    const int aOff0 = (wave * 32 + ln) * 64 + lg * 16;
    const int aOff1 = aOff0 + 16 * 64;                // second 16-row tile
    const int bOffC = 16384 + ln * 64 + lg * 16;

    v8f zero = {};
    v8f acc[2][NG];
#pragma unroll
    for (int t = 0; t < 2; ++t)
#pragma unroll
        for (int g = 0; g < NG; ++g) acc[t][g] = zero;

    // ---- prologue: stage tile 0 into buffer 0 ----
    ASYNC_B128(ldsBase + aDst, gA, "0");
    ASYNC_B128(ldsBase + aDst, gA, "16");
    ASYNC_B128(ldsBase + aDst, gA, "32");
    ASYNC_B128(ldsBase + aDst, gA, "48");
    ASYNC_B128(ldsBase + bDst, gB, "0");
    ASYNC_B128(ldsBase + bDst, gB, "16");

    unsigned curOff = 0u;                 // LDS byte offset of compute buffer
    unsigned nxtOff = TILE_BYTES;         // LDS byte offset of staging buffer
#pragma unroll 1
    for (int k = 0; k < KT; ++k) {
        if (k + 1 < KT) {
            // stage tile k+1 while tile k is consumed
            unsigned bb = ldsBase + nxtOff;
            int go = (k + 1) * 64;        // +32 bf16 K elements per tile
            const char* pa = gA + go;
            const char* pb = gB + go;
            ASYNC_B128(bb + aDst, pa, "0");
            ASYNC_B128(bb + aDst, pa, "16");
            ASYNC_B128(bb + aDst, pa, "32");
            ASYNC_B128(bb + aDst, pa, "48");
            ASYNC_B128(bb + bDst, pb, "0");
            ASYNC_B128(bb + bDst, pb, "16");
            // in-order completion: <=6 outstanding means tile k has landed
            asm volatile("s_wait_asynccnt 0x6" ::: "memory");
        } else {
            asm volatile("s_wait_asynccnt 0x0" ::: "memory");
        }
        __syncthreads();   // all waves' tile-k data visible; skew <= 1 iter

        const char* bp = (const char*)lds4 + curOff;
        union Frag { uint4 u[2]; v16bf v; };
        Frag aF0, aF1;
        aF0.u[0] = *(const uint4*)(bp + aOff0);
        aF0.u[1] = *(const uint4*)(bp + aOff0 + 32);
        aF1.u[0] = *(const uint4*)(bp + aOff1);
        aF1.u[1] = *(const uint4*)(bp + aOff1 + 32);

        // Manually software-pipelined gate chain with two named B fragments:
        // the overlapping live ranges force two physical register sets, so
        // each gate's ds loads issue two WMMA-pairs ahead of their use.
#define LOADB(f, g)                                                        \
        f.u[0] = *(const uint4*)(bp + bOffC + (g) * 1024);                 \
        f.u[1] = *(const uint4*)(bp + bOffC + (g) * 1024 + 32)
#define WMMA2(g, f)                                                        \
        acc[0][g] = __builtin_amdgcn_wmma_f32_16x16x32_bf16(               \
            false, aF0.v, false, f.v, (short)0, acc[0][g], false, false);  \
        acc[1][g] = __builtin_amdgcn_wmma_f32_16x16x32_bf16(               \
            false, aF1.v, false, f.v, (short)0, acc[1][g], false, false)

        Frag bF0, bF1;
        LOADB(bF0, 0);
        LOADB(bF1, 1);
        WMMA2(0, bF0);
        LOADB(bF0, 2);
        WMMA2(1, bF1);
        LOADB(bF1, 3);
        WMMA2(2, bF0);
        LOADB(bF0, 4);
        WMMA2(3, bF1);
        LOADB(bF1, 5);
        WMMA2(4, bF0);
        LOADB(bF0, 6);
        WMMA2(5, bF1);
        WMMA2(6, bF0);
#undef LOADB
#undef WMMA2

        // rotate triple buffer (branchless, loop-carried offsets)
        curOff = nxtOff;
        nxtOff = nxtOff + TILE_BYTES;
        if (nxtOff == NBUF * TILE_BYTES) nxtOff = 0u;
    }

    // ---------------- epilogue ----------------
    const int j = colBase + ln;
    const float bi  = bs[0 * H_SZ + j];
    const float bf  = bs[1 * H_SZ + j];
    const float bz  = bs[2 * H_SZ + j];
    const float bo  = bs[3 * H_SZ + j];
    const float bd  = bs[4 * H_SZ + j];
    const float bib = bs[5 * H_SZ + j];
    const float bfb = bs[6 * H_SZ + j];

    const size_t BH = (size_t)B_SZ * H_SZ;
#pragma unroll
    for (int t = 0; t < 2; ++t) {
#pragma unroll
        for (int r = 0; r < 8; ++r) {
            int m = r + lg * 8;                      // C/D layout: vgpr r -> M
            int b = rowBase + wave * 32 + t * 16 + m;
            size_t oi = (size_t)b * H_SZ + j;
            float ctv = ct[oi];

            float wi  = acc[t][0][r] + bi;
            float wf  = acc[t][1][r] + bf;
            float wz  = acc[t][2][r] + bz;
            float wo  = acc[t][3][r] + bo;
            float wd  = acc[t][4][r] + bd;
            float wib = acc[t][5][r] + bib;
            float wfb = acc[t][6][r] + bfb;

            float gi  = sigmoidf_(wi);
            float gf  = sigmoidf_(wf);
            float gz  = tanhf(wz);
            float go  = sigmoidf_(wo);
            float gib = sigmoidf_(wib);
            float gfb = sigmoidf_(wfb);

            float c   = gf * ctv + gi * gz;
            float h   = go * tanhf(c);
            float cb  = gfb * ctv + gib * gz;
            float dr  = fmaxf(wd, 0.0f) + log1pf(__expf(-fabsf(wd)));

            out[oi]          = h;
            out[BH + oi]     = c;
            out[2 * BH + oi] = cb;
            out[3 * BH + oi] = go;
            out[4 * BH + oi] = dr;
        }
    }
}

// ---------------------------------------------------------------------------
extern "C" void kernel_launch(void* const* d_in, const int* in_sizes, int n_in,
                              void* d_out, int out_size, void* d_ws, size_t ws_size,
                              hipStream_t stream) {
    const float* x  = (const float*)d_in[0];
    const float* ht = (const float*)d_in[1];
    const float* ct = (const float*)d_in[2];
    const float* Wx = (const float*)d_in[3];
    const float* bx = (const float*)d_in[4];
    const float* Wh = (const float*)d_in[5];
    const float* bh = (const float*)d_in[6];

    char*   ws  = (char*)d_ws;
    __bf16* Xc  = (__bf16*)(ws + XC_OFF);
    __bf16* Wc  = (__bf16*)(ws + WC_OFF);
    float*  bsm = (float*)(ws + BS_OFF);

    {   // pack [x|ht] -> bf16 [B][1024]
        int total4 = B_SZ * K_SZ / 4;
        pack_concat_bf16<<<(total4 + 255) / 256, 256, 0, stream>>>(x, ht, Xc, B_SZ);
    }
    {   // pack [Wx|Wh] -> bf16 [3584][1024]
        int total4 = NG * H_SZ * K_SZ / 4;
        pack_concat_bf16<<<(total4 + 255) / 256, 256, 0, stream>>>(Wx, Wh, Wc, NG * H_SZ);
    }
    bias_sum<<<(NG * H_SZ + 255) / 256, 256, 0, stream>>>(bx, bh, bsm);

    dim3 grid(B_SZ / 256, H_SZ / 16);
    ctlstm_wmma<<<grid, 256, 0, stream>>>(Xc, Wc, bsm, ct, (float*)d_out);
}
